// Barrier_Net_20392504721736
// MI455X (gfx1250) — matrix operand order repaired and checked by reading, not compiled, not run on hardware
//
#include <hip/hip_runtime.h>
#include <hip/hip_bf16.h>
#include <math.h>

typedef __attribute__((ext_vector_type(16))) _Float16 v16h;
typedef __attribute__((ext_vector_type(8)))  _Float16 v8h;
typedef __attribute__((ext_vector_type(8)))  float    v8f;

#define BATCH_N   4096
#define NBLK      66
#define INROW     264      // 66*4 floats per batch row
#define TILE_B    16
#define THREADS   128      // 4 waves of 32

#define HS_STRIDE 264      // padded row stride (halves) for H_sum  (multiple of 8, not of 128 dwords)
#define XS_STRIDE 72       // padded row stride (halves) for X tile

// workspace layout in halves: W2 packed [4 ntile][8 kstep][32 lane][16], then rho_w1 packed [16][2][32][16]
#define W2P_HALVES  (4 * 8 * 32 * 16)
#define W1RP_HALVES (16 * 2 * 32 * 16)

// ---------------------------------------------------------------------------
// Pack weights into the per-lane WMMA B-matrix layout (f16):
//   lane l covers col n = ntile*16 + (l&15), K rows kh*8 + j + (j>=8 ? 8 : 0) (kh = l>>4),
//   stored contiguously so the GEMM kernel does one 32-byte load per K-step.
// ---------------------------------------------------------------------------
__global__ void prep_weights(const float* __restrict__ phi_w2,   // (256,64)
                             const float* __restrict__ rho_w1,   // (64,256)
                             _Float16* __restrict__ wsp) {
  int tid = blockIdx.x * blockDim.x + threadIdx.x;
  if (tid < 1024) {                       // phi_w2 pack: ntile 0..3, kstep 0..7
    int lane = tid & 31, s = (tid >> 5) & 7;
    int n  = ((tid >> 8) * 16) + (lane & 15);
    int kh = lane >> 4;
    _Float16* dst = wsp + (size_t)tid * 16;
    for (int j = 0; j < 16; ++j)
      dst[j] = (_Float16)phi_w2[(s * 32 + kh * 8 + j + (j >= 8 ? 8 : 0)) * 64 + n];
  } else if (tid < 2048) {                // rho_w1 pack: ntile 0..15, kstep 0..1
    int t = tid - 1024;
    int lane = t & 31, s = (t >> 5) & 1;
    int n  = ((t >> 6) * 16) + (lane & 15);
    int kh = lane >> 4;
    _Float16* dst = wsp + W2P_HALVES + (size_t)t * 16;
    for (int j = 0; j < 16; ++j)
      dst[j] = (_Float16)rho_w1[(s * 32 + kh * 8 + j + (j >= 8 ? 8 : 0)) * 256 + n];
  }
}

// ---------------------------------------------------------------------------
// Fused main kernel: one block = 16 batch rows, 4 waves.
//  phase 1: H_sum[16][256] = sum_n relu(x_block @ phi_w1 + b1)   (VALU, K=4)
//  phase 2: X[16][64]      = H_sum @ phi_w2 + 66*phi_b2          (WMMA f16)
//  phase 3: h2 = relu(X @ rho_w1 + rho_b1); a_pre = h2 @ rho_w2  (WMMA f16 + LDS f32 atomics)
//  phase 4: a = 2*tanh(a_pre + rho_b2) + barrier(x)  -> d_out (pre-rescale)
// ---------------------------------------------------------------------------
__global__ __launch_bounds__(THREADS) void barrier_net_main(
    const float* __restrict__ x,
    const float* __restrict__ phi_w1, const float* __restrict__ phi_b1,
    const float* __restrict__ phi_b2,
    const float* __restrict__ rho_b1, const float* __restrict__ rho_w2,
    const float* __restrict__ rho_b2,
    const _Float16* __restrict__ W2p, const _Float16* __restrict__ W1rp,
    float* __restrict__ out)
{
  __shared__ __align__(16) float    xs[TILE_B * INROW];       // input tile
  __shared__ float                  W1s[4 * 256];             // phi_w1
  __shared__ float                  b1s[256];                 // phi_b1
  __shared__ __align__(16) _Float16 Hs[TILE_B * HS_STRIDE];   // H_sum (f16, padded rows)
  __shared__ __align__(16) _Float16 Xs[TILE_B * XS_STRIDE];   // X tile (f16, padded rows)
  __shared__ float                  accL[TILE_B * 2];         // a_pre accumulators

  const int tid = threadIdx.x;
  const int b0  = blockIdx.x * TILE_B;

  for (int i = tid; i < TILE_B * INROW; i += THREADS) xs[i]  = x[(size_t)b0 * INROW + i];
  for (int i = tid; i < 1024;           i += THREADS) W1s[i] = phi_w1[i];
  for (int i = tid; i < 256;            i += THREADS) b1s[i] = phi_b1[i];
  if (tid < TILE_B * 2) accL[tid] = 0.0f;
  __syncthreads();

  // ---- phase 1: per-row H_sum, thread owns (row, 32 hidden units) ----
  {
    const int row = tid >> 3;
    const int h0  = (tid & 7) * 32;
    float hacc[32];
    #pragma unroll
    for (int i = 0; i < 32; ++i) hacc[i] = 0.0f;
    for (int hb = 0; hb < 4; ++hb) {
      float w0[8], w1[8], w2c[8], w3[8], bb[8];
      #pragma unroll
      for (int i = 0; i < 8; ++i) {
        int h = h0 + hb * 8 + i;
        w0[i] = W1s[h]; w1[i] = W1s[256 + h]; w2c[i] = W1s[512 + h]; w3[i] = W1s[768 + h];
        bb[i] = b1s[h];
      }
      for (int n = 0; n < NBLK; ++n) {
        float s0 = xs[row * INROW + n * 4 + 0];
        float s1 = xs[row * INROW + n * 4 + 1];
        float s2 = xs[row * INROW + n * 4 + 2];
        float s3 = xs[row * INROW + n * 4 + 3];
        #pragma unroll
        for (int i = 0; i < 8; ++i) {
          float hv = fmaf(s0, w0[i], fmaf(s1, w1[i], fmaf(s2, w2c[i], fmaf(s3, w3[i], bb[i]))));
          hacc[hb * 8 + i] += fmaxf(hv, 0.0f);
        }
      }
    }
    #pragma unroll
    for (int i = 0; i < 32; ++i) Hs[row * HS_STRIDE + h0 + i] = (_Float16)hacc[i];
  }
  __syncthreads();

  const int lane  = tid & 31;
  const int wv    = tid >> 5;       // wave id 0..3
  const int rowA  = lane & 15;      // A-matrix row / C-matrix col index
  const int kh    = lane >> 4;      // half-wave selector for A layout
  const int rbase = kh * 8;         // C-matrix row base for this lane

  // ---- phase 2: X = H_sum @ phi_w2 + 66*phi_b2  (wave wv owns cols 16*wv..) ----
  {
    v8f c = {};
    #pragma unroll
    for (int s = 0; s < 8; ++s) {
      v8h lo = *(const v8h*)&Hs[rowA * HS_STRIDE + s * 32 + kh * 8];
      v8h hi = *(const v8h*)&Hs[rowA * HS_STRIDE + s * 32 + 16 + kh * 8];
      v16h a;
      #pragma unroll
      for (int i = 0; i < 8; ++i) { a[i] = lo[i]; a[8 + i] = hi[i]; }
      v16h b = *(const v16h*)(W2p + ((size_t)((wv * 8 + s) * 32 + lane)) * 16);
      c = __builtin_amdgcn_wmma_f32_16x16x32_f16(false, a, false, b, (short)0, c, false, false);
    }
    int   nX   = wv * 16 + rowA;
    float bias = 66.0f * phi_b2[nX];
    #pragma unroll
    for (int i = 0; i < 8; ++i)
      Xs[(rbase + i) * XS_STRIDE + nX] = (_Float16)(c[i] + bias);
  }
  __syncthreads();

  // ---- phase 3: h2 = relu(X @ rho_w1 + b); fold into a_pre = h2 @ rho_w2 ----
  {
    float p0a[8], p1a[8];
    #pragma unroll
    for (int i = 0; i < 8; ++i) { p0a[i] = 0.0f; p1a[i] = 0.0f; }
    #pragma unroll
    for (int q = 0; q < 4; ++q) {
      int nt = wv * 4 + q;
      v8f c2 = {};
      #pragma unroll
      for (int s = 0; s < 2; ++s) {
        v8h lo = *(const v8h*)&Xs[rowA * XS_STRIDE + s * 32 + kh * 8];
        v8h hi = *(const v8h*)&Xs[rowA * XS_STRIDE + s * 32 + 16 + kh * 8];
        v16h a;
        #pragma unroll
        for (int i = 0; i < 8; ++i) { a[i] = lo[i]; a[8 + i] = hi[i]; }
        v16h b = *(const v16h*)(W1rp + ((size_t)((nt * 2 + s) * 32 + lane)) * 16);
        c2 = __builtin_amdgcn_wmma_f32_16x16x32_f16(false, a, false, b, (short)0, c2, false, false);
      }
      int   n2  = nt * 16 + rowA;
      float br  = rho_b1[n2];
      float rw0 = rho_w2[n2 * 2 + 0];
      float rw1 = rho_w2[n2 * 2 + 1];
      #pragma unroll
      for (int i = 0; i < 8; ++i) {
        float h2 = fmaxf(c2[i] + br, 0.0f);
        p0a[i] = fmaf(h2, rw0, p0a[i]);
        p1a[i] = fmaf(h2, rw1, p1a[i]);
      }
    }
    #pragma unroll
    for (int i = 0; i < 8; ++i) {
      atomicAdd(&accL[(rbase + i) * 2 + 0], p0a[i]);
      atomicAdd(&accL[(rbase + i) * 2 + 1], p1a[i]);
    }
  }
  __syncthreads();

  // ---- phase 4: tanh head + barrier term, write pre-rescale outputs ----
  if (tid < TILE_B * 2) {
    int r = tid & 15, j = tid >> 4;
    float av = 2.0f * tanhf(accL[r * 2 + j] + rho_b2[j]);
    float bsum = 0.0f;
    for (int n = 0; n < 64; ++n) {
      float px = xs[r * INROW + (2 + n) * 4 + 0];
      float py = xs[r * INROW + (2 + n) * 4 + 1];
      float d  = sqrtf(px * px + py * py) - 0.15f;
      float ps = j ? py : px;
      bsum -= ps / (d * d);
    }
    out[(size_t)(b0 + r) * 2 + j] = av + bsum;
  }
}

// ---------------------------------------------------------------------------
// Global max over all 8192 outputs, then conditional in-place rescale.
// Single block -> no cross-block race.
// ---------------------------------------------------------------------------
__global__ void finalize_scale(float* __restrict__ out) {
  __shared__ float red[256];
  float m = -INFINITY;
  for (int i = threadIdx.x; i < BATCH_N * 2; i += 256) m = fmaxf(m, out[i]);
  red[threadIdx.x] = m;
  __syncthreads();
  for (int st = 128; st > 0; st >>= 1) {
    if (threadIdx.x < st) red[threadIdx.x] = fmaxf(red[threadIdx.x], red[threadIdx.x + st]);
    __syncthreads();
  }
  float scale = 2.0f / red[0];
  if (scale < 1.0f) {
    for (int i = threadIdx.x; i < BATCH_N * 2; i += 256) out[i] *= scale;
  }
}

extern "C" void kernel_launch(void* const* d_in, const int* in_sizes, int n_in,
                              void* d_out, int out_size, void* d_ws, size_t ws_size,
                              hipStream_t stream) {
  const float* x      = (const float*)d_in[0];
  const float* phi_w1 = (const float*)d_in[1];
  const float* phi_b1 = (const float*)d_in[2];
  const float* phi_w2 = (const float*)d_in[3];
  const float* phi_b2 = (const float*)d_in[4];
  const float* rho_w1 = (const float*)d_in[5];
  const float* rho_b1 = (const float*)d_in[6];
  const float* rho_w2 = (const float*)d_in[7];
  const float* rho_b2 = (const float*)d_in[8];
  (void)in_sizes; (void)n_in; (void)out_size; (void)ws_size;

  _Float16* wsp = (_Float16*)d_ws;

  prep_weights<<<8, 256, 0, stream>>>(phi_w2, rho_w1, wsp);
  barrier_net_main<<<BATCH_N / TILE_B, THREADS, 0, stream>>>(
      x, phi_w1, phi_b1, phi_b2, rho_b1, rho_w2, rho_b2,
      wsp, wsp + W2P_HALVES, (float*)d_out);
  finalize_scale<<<1, 256, 0, stream>>>((float*)d_out);
}
